// FlashNgramModel_48421461295452
// MI455X (gfx1250) — compile-verified, matrix operand order checked by reading, not run on hardware
//
#include <hip/hip_runtime.h>
#include <hip/hip_bf16.h>
#include <math.h>

// Problem constants (match the reference)
#define T_TOK   1024
#define H_DIM   1024
#define I_DIM   512
#define N_EXP   32
#define N_ZERO  8
#define N_ROUTE 40   // E + Z
#define TOPK    4
#define ROUTED_SCALING 1.0f

#define TILE_M  32                                  // tokens per workgroup tile

typedef __attribute__((ext_vector_type(16))) __bf16 v16bf;
typedef __attribute__((ext_vector_type(8)))  __bf16 v8bf;
typedef __attribute__((ext_vector_type(8)))  float  v8f;

// ---------------- workspace layout (bytes) ----------------
#define OFF_HBF16   (0)                                       // T*H bf16 = 2 MB
#define OFF_LOGITS  (OFF_HBF16  + T_TOK * H_DIM * 2)          // T*40 f32
#define OFF_TOPKI   (OFF_LOGITS + T_TOK * N_ROUTE * 4)        // T*4 i32
#define OFF_TOPKW   (OFF_TOPKI  + T_TOK * TOPK * 4)           // T*4 f32
#define OFF_ZW      (OFF_TOPKW  + T_TOK * TOPK * 4)           // T f32
#define OFF_CNT     (OFF_ZW     + T_TOK * 4)                  // E i32
#define OFF_RANK    (OFF_CNT    + 256)                        // T*4 i32
#define OFF_POFF    (OFF_RANK   + T_TOK * TOPK * 4)           // (E+1) i32
#define OFF_TOTAL   (OFF_POFF   + 256)                        // 1 i32
#define OFF_PAIRS   (OFF_TOTAL  + 256)                        // pair list
#define PAIR_CAP    5120                                      // 4096 + 32*31, rounded
#define OFF_SLOT    (OFF_PAIRS  + PAIR_CAP * 4)               // T*4*H f32 = 16 MB

#define MAX_TILES   ((T_TOK * TOPK) / TILE_M + N_EXP)         // 160

// =========================================================
// 0) init: zero expert counters, poison the pair list
// =========================================================
__global__ void moe_init_kernel(int* counts, int* pairs) {
    int g = blockIdx.x * blockDim.x + threadIdx.x;
    if (g < PAIR_CAP) pairs[g] = -1;
    if (g < N_EXP)    counts[g] = 0;
}

// =========================================================
// 1) hidden fp32 -> bf16 (A operand for all WMMA GEMMs)
// =========================================================
__global__ void cvt_hidden_bf16_kernel(const float* __restrict__ hid,
                                       __bf16* __restrict__ hbf) {
    int g = blockIdx.x * blockDim.x + threadIdx.x;      // T*H/8 threads
    if (g >= (T_TOK * H_DIM) / 8) return;
    const float* src = hid + (size_t)g * 8;
    v8bf o;
#pragma unroll
    for (int j = 0; j < 8; ++j) o[j] = (__bf16)src[j];
    *reinterpret_cast<v8bf*>(hbf + (size_t)g * 8) = o;
}

// =========================================================
// 2) router logits [T,40] = hidden @ router_w^T   (fp32, tiny)
// =========================================================
__global__ void router_logits_kernel(const float* __restrict__ hid,
                                     const float* __restrict__ rw,
                                     float* __restrict__ logits) {
    int g = blockIdx.x * blockDim.x + threadIdx.x;      // T*40 threads
    if (g >= T_TOK * N_ROUTE) return;
    int t = g / N_ROUTE, e = g % N_ROUTE;
    const float* hr = hid + (size_t)t * H_DIM;
    const float* wr = rw  + (size_t)e * H_DIM;
    float s = 0.f;
#pragma unroll 4
    for (int k = 0; k < H_DIM; k += 4) {
        float4 a = *reinterpret_cast<const float4*>(hr + k);
        float4 b = *reinterpret_cast<const float4*>(wr + k);
        s += a.x * b.x + a.y * b.y + a.z * b.z + a.w * b.w;
    }
    logits[(size_t)t * N_ROUTE + e] = s;
}

// =========================================================
// 3) softmax + top-4 selection + zero-expert weight + rank
// =========================================================
__global__ void router_topk_kernel(const float* __restrict__ logits,
                                   const float* __restrict__ cbias,
                                   int* __restrict__ topkIdx,
                                   float* __restrict__ topkW,
                                   float* __restrict__ zeroW,
                                   int* __restrict__ counts,
                                   int* __restrict__ rank) {
    int t = blockIdx.x * blockDim.x + threadIdx.x;
    if (t >= T_TOK) return;
    float lg[N_ROUTE], sc[N_ROUTE];
    float mx = -3.4e38f;
#pragma unroll
    for (int j = 0; j < N_ROUTE; ++j) { lg[j] = logits[(size_t)t * N_ROUTE + j]; mx = fmaxf(mx, lg[j]); }
    float den = 0.f;
#pragma unroll
    for (int j = 0; j < N_ROUTE; ++j) { sc[j] = expf(lg[j] - mx); den += sc[j]; }
    float inv = 1.f / den;
#pragma unroll
    for (int j = 0; j < N_ROUTE; ++j) sc[j] *= inv;

    unsigned long long used = 0ull;
    float zw = 0.f;
    for (int s = 0; s < TOPK; ++s) {
        int   best = -1;
        float bv   = -3.4e38f;
        for (int j = 0; j < N_ROUTE; ++j) {
            if (used & (1ull << j)) continue;
            float v = sc[j] + cbias[j];
            if (v > bv) { bv = v; best = j; }
        }
        used |= (1ull << best);
        float w = sc[best] * ROUTED_SCALING;
        topkIdx[t * TOPK + s] = best;
        topkW  [t * TOPK + s] = w;
        if (best < N_EXP) {
            rank[t * TOPK + s] = atomicAdd(&counts[best], 1);
        } else {
            rank[t * TOPK + s] = -1;
            zw += w;
        }
    }
    zeroW[t] = zw;
}

// =========================================================
// 4) exclusive scan of counts, padded to TILE_M-token tiles
// =========================================================
__global__ void scan_offsets_kernel(const int* __restrict__ counts,
                                    int* __restrict__ poff,
                                    int* __restrict__ total) {
    if (threadIdx.x == 0 && blockIdx.x == 0) {
        int acc = 0;
        for (int e = 0; e < N_EXP; ++e) {
            poff[e] = acc;
            acc += ((counts[e] + TILE_M - 1) / TILE_M) * TILE_M;
        }
        poff[N_EXP] = acc;
        *total = acc;
    }
}

// =========================================================
// 5) scatter (token,slot) pairs into expert-sorted order
// =========================================================
__global__ void scatter_pairs_kernel(const int* __restrict__ topkIdx,
                                     const int* __restrict__ rank,
                                     const int* __restrict__ poff,
                                     int* __restrict__ pairs) {
    int g = blockIdx.x * blockDim.x + threadIdx.x;    // T*4 threads
    if (g >= T_TOK * TOPK) return;
    int e = topkIdx[g];
    if (e < N_EXP) {
        int t = g >> 2, s = g & 3;
        pairs[poff[e] + rank[g]] = (t << 2) | s;
    }
}

// =========================================================
// fragment helpers (wave32 WMMA bf16, 16x16x32)
// =========================================================
// A/B 16-bit fragment: lanes 0-15 hold K = {0..7,16..23}, lanes 16-31
// hold K = {8..15,24..31}; M (or N) = lane & 15.
__device__ inline v16bf frag_from_lds(const __bf16* __restrict__ row0,
                                      int stride, int lane, int kk) {
    int m     = lane & 15;
    int halfk = (lane >> 4) * 8;
    const __bf16* p = row0 + (size_t)m * stride + kk + halfk;
    v8bf lo = *reinterpret_cast<const v8bf*>(p);
    v8bf hi = *reinterpret_cast<const v8bf*>(p + 16);
    return __builtin_shufflevector(lo, hi, 0,1,2,3,4,5,6,7,8,9,10,11,12,13,14,15);
}

__device__ inline v16bf frag_from_global_f32(const float* __restrict__ base,
                                             int ldb, int lane, int kk) {
    int n     = lane & 15;
    int halfk = (lane >> 4) * 8;
    const float* p = base + (size_t)n * ldb + kk + halfk;
    v8bf lo, hi;
#pragma unroll
    for (int j = 0; j < 8; ++j) {
        lo[j] = (__bf16)p[j];
        hi[j] = (__bf16)p[j + 16];
    }
    return __builtin_shufflevector(lo, hi, 0,1,2,3,4,5,6,7,8,9,10,11,12,13,14,15);
}

// =========================================================
// 6) grouped expert GEMM: one 32-token tile per workgroup.
//    8 waves; each wave owns 128 output columns (8 WMMA N-tiles)
//    x 2 M-subtiles -> each B fragment feeds 2 WMMAs.
//    GEMM1 (32x2I) -> silu-gate (up half swapped via LDS) ->
//    GEMM2 (32xH) -> weighted scatter to per-slot strips.
// =========================================================
#define A_STRIDE   1032          // bf16 elems, multiple of 8 -> 16B aligned frags
#define HM_STRIDE  520
#define SM_REGION0 (TILE_M * A_STRIDE * 2)            // A (66 KB) ... also >= UP (64 KB)
#define SM_HMID    (SM_REGION0)                       // 32*520*2 = 33280
#define SM_STAGE   (SM_HMID + TILE_M * HM_STRIDE * 2)
#define SM_BYTES   (SM_STAGE + 512)

__global__ __launch_bounds__(256, 1)
void moe_expert_gemm_kernel(const __bf16* __restrict__ hbf,
                            const float* __restrict__ wgu,
                            const float* __restrict__ wdn,
                            const int*   __restrict__ pairs,
                            const int*   __restrict__ poff,
                            const int*   __restrict__ total,
                            const float* __restrict__ topkW,
                            float* __restrict__ slotbuf) {
    __shared__ __align__(16) unsigned char smem[SM_BYTES];
    __bf16* ldsA  = reinterpret_cast<__bf16*>(smem);            // [32][A_STRIDE]
    float*  ldsUP = reinterpret_cast<float*>(smem);             // [32][512] (aliases A)
    __bf16* ldsHM = reinterpret_cast<__bf16*>(smem + SM_HMID);  // [32][HM_STRIDE]
    int*    sTok  = reinterpret_cast<int*>(smem + SM_STAGE);    // [32]
    int*    sSlot = sTok + TILE_M;                              // [32]
    float*  sW    = reinterpret_cast<float*>(sSlot + TILE_M);   // [32]

    const int base = blockIdx.x * TILE_M;
    if (base >= *total) return;                 // uniform early-exit

    const int tid  = threadIdx.x;
    const int lane = tid & 31;
    const int wid  = tid >> 5;                  // 8 waves

    // tile -> expert (binary search over padded offsets, 33 entries)
    int e = 0;
#pragma unroll
    for (int step = 16; step >= 1; step >>= 1)
        if (e + step <= N_EXP - 1 && poff[e + step] <= base) e += step;

    // stage pair metadata
    if (tid < TILE_M) {
        int p = pairs[base + tid];
        sTok [tid] = (p >= 0) ? (p >> 2) : -1;
        sSlot[tid] = (p >= 0) ? (p & 3) : 0;
        sW   [tid] = (p >= 0) ? topkW[p] : 0.f;
    }
    __syncthreads();

    // stage gathered A rows: 32 tokens x 1024 bf16 (64 KB)
    {
        int row  = tid >> 3;                    // 32 rows, 8 threads/row
        int col0 = (tid & 7) * 128;             // 128 bf16 = 256 B per thread
        int tok  = sTok[row];
        uint4* dst = reinterpret_cast<uint4*>(ldsA + (size_t)row * A_STRIDE + col0);
        if (tok >= 0) {
            const uint4* src = reinterpret_cast<const uint4*>(hbf + (size_t)tok * H_DIM + col0);
#pragma unroll
            for (int j = 0; j < 16; ++j) dst[j] = src[j];
        } else {
            uint4 z = {0u, 0u, 0u, 0u};
#pragma unroll
            for (int j = 0; j < 16; ++j) dst[j] = z;
        }
    }
    __syncthreads();

    // ---------- GEMM1: gu[32, 2I] = A[32,H] @ w_gate_up[e]^T ----------
    const float* wgu_e = wgu + (size_t)e * (2 * I_DIM) * H_DIM;
    v8f acc[2][8];
#pragma unroll
    for (int mt = 0; mt < 2; ++mt)
#pragma unroll
        for (int t = 0; t < 8; ++t) { v8f z = {}; acc[mt][t] = z; }

    for (int kk = 0; kk < H_DIM; kk += 32) {
        v16bf a0 = frag_from_lds(ldsA,                  A_STRIDE, lane, kk);
        v16bf a1 = frag_from_lds(ldsA + 16 * A_STRIDE,  A_STRIDE, lane, kk);
#pragma unroll
        for (int t = 0; t < 8; ++t) {
            const float* bbase = wgu_e + (size_t)(wid * 128 + t * 16) * H_DIM;
            __builtin_prefetch(bbase + kk + 128, 0, 0);
            v16bf b = frag_from_global_f32(bbase, H_DIM, lane, kk);
            acc[0][t] = __builtin_amdgcn_wmma_f32_16x16x32_bf16(
                            false, a0, false, b, (short)0, acc[0][t], false, false);
            acc[1][t] = __builtin_amdgcn_wmma_f32_16x16x32_bf16(
                            false, a1, false, b, (short)0, acc[1][t], false, false);
        }
    }
    __syncthreads();     // done reading A -> safe to overwrite union with UP

    // waves 4..7 own gu columns [512,1024) = "up": park them in LDS
    if (wid >= 4) {
#pragma unroll
        for (int mt = 0; mt < 2; ++mt)
#pragma unroll
            for (int t = 0; t < 8; ++t) {
                int c = (wid - 4) * 128 + t * 16 + (lane & 15);   // up column
#pragma unroll
                for (int r = 0; r < 8; ++r) {
                    int m = mt * 16 + r + 8 * (lane >> 4);
                    ldsUP[(size_t)m * I_DIM + c] = acc[mt][t][r];
                }
            }
    }
    __syncthreads();

    // waves 0..3 own gu columns [0,512) = "gate": silu-gate into hmid (bf16)
    if (wid < 4) {
#pragma unroll
        for (int mt = 0; mt < 2; ++mt)
#pragma unroll
            for (int t = 0; t < 8; ++t) {
                int c = wid * 128 + t * 16 + (lane & 15);         // gate column
#pragma unroll
                for (int r = 0; r < 8; ++r) {
                    int m = mt * 16 + r + 8 * (lane >> 4);
                    float g = acc[mt][t][r];
                    float u = ldsUP[(size_t)m * I_DIM + c];
                    float s = g / (1.f + expf(-g));
                    ldsHM[(size_t)m * HM_STRIDE + c] = (__bf16)(s * u);
                }
            }
    }
    __syncthreads();

    // ---------- GEMM2: y[32, H] = hmid[32,I] @ w_down[e]^T ----------
    const float* wdn_e = wdn + (size_t)e * H_DIM * I_DIM;
    v8f acc2[2][8];
#pragma unroll
    for (int mt = 0; mt < 2; ++mt)
#pragma unroll
        for (int t = 0; t < 8; ++t) { v8f z = {}; acc2[mt][t] = z; }

    for (int kk = 0; kk < I_DIM; kk += 32) {
        v16bf a0 = frag_from_lds(ldsHM,                   HM_STRIDE, lane, kk);
        v16bf a1 = frag_from_lds(ldsHM + 16 * HM_STRIDE,  HM_STRIDE, lane, kk);
#pragma unroll
        for (int t = 0; t < 8; ++t) {
            const float* bbase = wdn_e + (size_t)(wid * 128 + t * 16) * I_DIM;
            __builtin_prefetch(bbase + kk + 128, 0, 0);
            v16bf b = frag_from_global_f32(bbase, I_DIM, lane, kk);
            acc2[0][t] = __builtin_amdgcn_wmma_f32_16x16x32_bf16(
                             false, a0, false, b, (short)0, acc2[0][t], false, false);
            acc2[1][t] = __builtin_amdgcn_wmma_f32_16x16x32_bf16(
                             false, a1, false, b, (short)0, acc2[1][t], false, false);
        }
    }

    // weighted scatter to per-(token,slot) strips
#pragma unroll
    for (int mt = 0; mt < 2; ++mt)
#pragma unroll
        for (int t = 0; t < 8; ++t) {
            int h = wid * 128 + t * 16 + (lane & 15);
#pragma unroll
            for (int r = 0; r < 8; ++r) {
                int m   = mt * 16 + r + 8 * (lane >> 4);
                int tok = sTok[m];
                if (tok >= 0) {
                    size_t o = (((size_t)tok * TOPK + sSlot[m]) * H_DIM) + h;
                    slotbuf[o] = acc2[mt][t][r] * sW[m];
                }
            }
        }
}

// =========================================================
// 7) combine: out = zero_w * hidden + sum of routed slots
// =========================================================
__global__ void combine_kernel(const float* __restrict__ hid,
                               const float* __restrict__ slotbuf,
                               const int*   __restrict__ topkIdx,
                               const float* __restrict__ zeroW,
                               float* __restrict__ out) {
    int g = blockIdx.x * blockDim.x + threadIdx.x;      // T*H/4 threads
    if (g >= T_TOK * (H_DIM / 4)) return;
    int t  = g / (H_DIM / 4);
    int h4 = (g % (H_DIM / 4)) * 4;
    float4 hv = *reinterpret_cast<const float4*>(hid + (size_t)t * H_DIM + h4);
    float  zw = zeroW[t];
    float4 o = {zw * hv.x, zw * hv.y, zw * hv.z, zw * hv.w};
#pragma unroll
    for (int s = 0; s < TOPK; ++s) {
        if (topkIdx[t * TOPK + s] < N_EXP) {
            const float4 sv = *reinterpret_cast<const float4*>(
                slotbuf + (((size_t)t * TOPK + s) * H_DIM) + h4);
            o.x += sv.x; o.y += sv.y; o.z += sv.z; o.w += sv.w;
        }
    }
    *reinterpret_cast<float4*>(out + (size_t)t * H_DIM + h4) = o;
}

// =========================================================
extern "C" void kernel_launch(void* const* d_in, const int* in_sizes, int n_in,
                              void* d_out, int out_size, void* d_ws, size_t ws_size,
                              hipStream_t stream) {
    const float* hid  = (const float*)d_in[0];   // [T, H]
    const float* rw   = (const float*)d_in[1];   // [E+Z, H]
    const float* cb   = (const float*)d_in[2];   // [E+Z]
    const float* wgu  = (const float*)d_in[3];   // [E, 2I, H]
    const float* wdn  = (const float*)d_in[4];   // [E, H, I]
    float*       out  = (float*)d_out;           // [T, H]

    unsigned char* ws = (unsigned char*)d_ws;
    __bf16* hbf     = (__bf16*)(ws + OFF_HBF16);
    float*  logits  = (float*) (ws + OFF_LOGITS);
    int*    topkIdx = (int*)   (ws + OFF_TOPKI);
    float*  topkW   = (float*) (ws + OFF_TOPKW);
    float*  zeroW   = (float*) (ws + OFF_ZW);
    int*    counts  = (int*)   (ws + OFF_CNT);
    int*    rank    = (int*)   (ws + OFF_RANK);
    int*    poff    = (int*)   (ws + OFF_POFF);
    int*    total   = (int*)   (ws + OFF_TOTAL);
    int*    pairs   = (int*)   (ws + OFF_PAIRS);
    float*  slotbuf = (float*) (ws + OFF_SLOT);

    moe_init_kernel<<<(PAIR_CAP + 255) / 256, 256, 0, stream>>>(counts, pairs);

    cvt_hidden_bf16_kernel<<<(T_TOK * H_DIM / 8 + 255) / 256, 256, 0, stream>>>(hid, hbf);

    router_logits_kernel<<<(T_TOK * N_ROUTE + 255) / 256, 256, 0, stream>>>(hid, rw, logits);

    router_topk_kernel<<<(T_TOK + 255) / 256, 256, 0, stream>>>(
        logits, cb, topkIdx, topkW, zeroW, counts, rank);

    scan_offsets_kernel<<<1, 32, 0, stream>>>(counts, poff, total);

    scatter_pairs_kernel<<<(T_TOK * TOPK + 255) / 256, 256, 0, stream>>>(
        topkIdx, rank, poff, pairs);

    moe_expert_gemm_kernel<<<MAX_TILES, 256, 0, stream>>>(
        hbf, wgu, wdn, pairs, poff, total, topkW, slotbuf);

    combine_kernel<<<(T_TOK * (H_DIM / 4) + 255) / 256, 256, 0, stream>>>(
        hid, slotbuf, topkIdx, zeroW, out);
}
